// NanoGPT_53343493816398
// MI455X (gfx1250) — matrix operand order repaired and checked by reading, not compiled, not run on hardware
//
#include <hip/hip_runtime.h>
#include <hip/hip_bf16.h>
#include <cstdint>

typedef _Float16 v16h __attribute__((ext_vector_type(16)));
typedef _Float16 v8h  __attribute__((ext_vector_type(8)));
typedef float    v8f  __attribute__((ext_vector_type(8)));

// D = A(16x32 f16) x B(32x16 f16) + C(16x16 f32), wave32
__device__ inline v8f wmma_f16(v16h a, v16h b, v8f c) {
    return __builtin_amdgcn_wmma_f32_16x16x32_f16(
        /*neg_a=*/false, a, /*neg_b=*/false, b,
        /*c_mod=*/(short)0, c, /*reuse_a=*/false, /*reuse_b=*/false);
}

// ---------------------------------------------------------------------------
// Tensor Data Mover: 2D f16 tile (tile_d0 elems x tile_d1 rows) from global
// (row stride = stride_elems) into contiguous LDS at lds_addr.
// D# layout per CDNA5 ISA.
#if __has_builtin(__builtin_amdgcn_tensor_load_to_lds)
#define GPT_HAVE_TDM 1
typedef unsigned int tdm_u32x4 __attribute__((ext_vector_type(4)));
typedef int          tdm_i32x8 __attribute__((ext_vector_type(8)));
typedef int          tdm_i32x4 __attribute__((ext_vector_type(4)));

__device__ inline void tdm_load_tile_f16(unsigned lds_addr, const void* gaddr,
                                         unsigned tensor_d0, unsigned tensor_d1,
                                         unsigned tile_d0, unsigned tile_d1,
                                         unsigned stride_elems) {
    unsigned long long ga = (unsigned long long)(uintptr_t)gaddr;
    tdm_u32x4 g0;
    g0[0] = 1u;                                   // count=1, user descriptor
    g0[1] = lds_addr;                             // LDS byte address
    g0[2] = (unsigned)(ga & 0xffffffffu);         // global_addr[31:0]
    g0[3] = (unsigned)((ga >> 32) & 0x01ffffffu)  // global_addr[56:32]
            | (2u << 30);                         // type=2 ("image")
    tdm_i32x8 g1;
    g1[0] = (int)(1u << 16);                      // data_size=1 (2 bytes)
    g1[1] = (int)((tensor_d0 & 0xffffu) << 16);   // tensor_dim0[15:0] @ bits63:48
    g1[2] = (int)((tensor_d0 >> 16) | ((tensor_d1 & 0xffffu) << 16));
    g1[3] = (int)((tensor_d1 >> 16) | (tile_d0 << 16));   // tile_dim0 @127:112
    g1[4] = (int)(tile_d1 & 0xffffu);             // tile_dim1; tile_dim2=0
    g1[5] = (int)stride_elems;                    // tensor_dim0_stride[31:0]
    g1[6] = 0;                                    // stride hi / dim1_stride lo
    g1[7] = 0;
    tdm_i32x4 z4;
    z4[0] = z4[1] = z4[2] = z4[3] = 0;
#if defined(__clang_major__) && (__clang_major__ >= 23)
    tdm_i32x8 z8;
    for (int i = 0; i < 8; ++i) z8[i] = 0;
    __builtin_amdgcn_tensor_load_to_lds(g0, g1, z4, z4, z8, 0);
#else
    __builtin_amdgcn_tensor_load_to_lds(g0, g1, z4, z4, 0);
#endif
}
#endif

// ---------------------------------------------------------------------------
// Embedding: x[b*T+t, :] = wte[idx[b,t], :] + wpe[t, :]
__global__ void embed_kernel(const int* __restrict__ idx,
                             const float* __restrict__ wte,
                             const float* __restrict__ wpe,
                             float* __restrict__ x) {
    int row = blockIdx.x;            // b*1024 + t
    int t   = row & 1023;
    int tok = idx[row];
    int i   = threadIdx.x;           // 256 threads, 2 elems each (E=512)
    const float* wr = wte + (size_t)tok * 512;
    const float* pr = wpe + (size_t)t * 512;
    float* xr = x + (size_t)row * 512;
    xr[i]       = wr[i]       + pr[i];
    xr[i + 256] = wr[i + 256] + pr[i + 256];
}

// ---------------------------------------------------------------------------
// LayerNorm over E=512, output f16. grid = 2048 rows, block = 256
__global__ void ln_kernel(const float* __restrict__ x,
                          const float* __restrict__ g,
                          const float* __restrict__ bb,
                          _Float16* __restrict__ out) {
    __shared__ float red[256];
    int row = blockIdx.x;
    int t = threadIdx.x;
    const float* xr = x + (size_t)row * 512;
    float a0 = xr[t], a1 = xr[t + 256];
    red[t] = a0 + a1;
    __syncthreads();
    for (int s = 128; s > 0; s >>= 1) {
        if (t < s) red[t] += red[t + s];
        __syncthreads();
    }
    float mean = red[0] * (1.0f / 512.0f);
    __syncthreads();
    float d0 = a0 - mean, d1 = a1 - mean;
    red[t] = d0 * d0 + d1 * d1;
    __syncthreads();
    for (int s = 128; s > 0; s >>= 1) {
        if (t < s) red[t] += red[t + s];
        __syncthreads();
    }
    float rinv = rsqrtf(red[0] * (1.0f / 512.0f) + 1e-5f);
    _Float16* o = out + (size_t)row * 512;
    o[t]       = (_Float16)(d0 * rinv * g[t]       + bb[t]);
    o[t + 256] = (_Float16)(d1 * rinv * g[t + 256] + bb[t + 256]);
}

// ---------------------------------------------------------------------------
// Tiled WMMA GEMM: out[M,N] = A[M,K](f16) * B(f32, converted) (+bias)(+res)(GELU)
// BT: B given as [N,K] (row = n), else [K,N]. Block = 128 threads (4 waves),
// tile 64x64, K-step 32. Each wave owns a 32x32 quadrant (2x2 WMMA subtiles).
// A tile is DMA'd into LDS by the Tensor Data Mover (wave 0, TENSORcnt),
// overlapped with the B f32->f16 conversion staging done by all waves.
template <bool BT, bool DOGELU, bool OUTF16, bool RES, bool BIAS>
__global__ __launch_bounds__(128)
void gemm_kernel(const _Float16* __restrict__ A, int lda,
                 const float* __restrict__ Bw, int ldb,
                 const float* __restrict__ bias,
                 const float* __restrict__ res,
                 float* __restrict__ out32,
                 _Float16* __restrict__ out16,
                 int ldo, int M, int N, int K) {
    __shared__ _Float16 As[64 * 32];   // [m][k]
    __shared__ _Float16 Bt[64 * 32];   // [n][k]  (transposed in LDS)
    (void)M;
    int t    = threadIdx.x;
    int wid  = t >> 5;
    int lane = t & 31;
    int lh   = lane >> 4;      // half-wave select
    int lm   = lane & 15;      // lane within half
    int wm   = wid >> 1, wn = wid & 1;
    int m0 = blockIdx.y * 64, n0 = blockIdx.x * 64;

    v8f acc[2][2];
    for (int i = 0; i < 2; ++i)
        for (int j = 0; j < 2; ++j)
            for (int e = 0; e < 8; ++e) acc[i][j][e] = 0.0f;

#if GPT_HAVE_TDM
    unsigned lds_a = (unsigned)(uintptr_t)(&As[0]);
#endif

    for (int k0 = 0; k0 < K; k0 += 32) {
        // --- stage A tile (64 rows x 32 halves) ---
#if GPT_HAVE_TDM
        if (wid == 0) {
            tdm_load_tile_f16(lds_a, A + (size_t)m0 * lda + k0,
                              /*tensor_d0=*/(unsigned)lda, /*tensor_d1=*/64u,
                              /*tile_d0=*/32u, /*tile_d1=*/64u,
                              /*stride=*/(unsigned)lda);
        }
#else
        {
            int r = t >> 1, kc = t & 1;
            *(v16h*)&As[r * 32 + kc * 16] =
                *(const v16h*)(A + (size_t)(m0 + r) * lda + k0 + kc * 16);
        }
#endif
        // --- stage B tile into [n][k] with f32->f16 conversion ---
        if (BT) {
            int n = t >> 1, kc = t & 1;
            const float* src = Bw + (size_t)(n0 + n) * ldb + k0 + kc * 16;
            v16h val;
            if (n0 + n < N) {
                for (int e = 0; e < 16; ++e) val[e] = (_Float16)src[e];
            } else {
                for (int e = 0; e < 16; ++e) val[e] = (_Float16)0.0f;
            }
            *(v16h*)&Bt[n * 32 + kc * 16] = val;
        } else {
            int k = t >> 2, nc = t & 3;
            const float* src = Bw + (size_t)(k0 + k) * ldb + n0 + nc * 16;
            for (int e = 0; e < 16; ++e) {
                int col = n0 + nc * 16 + e;
                Bt[(nc * 16 + e) * 32 + k] =
                    (col < N) ? (_Float16)src[e] : (_Float16)0.0f;
            }
        }
#if GPT_HAVE_TDM
        if (wid == 0) __builtin_amdgcn_s_wait_tensorcnt(0);
#endif
        __syncthreads();

        // --- fragments ---
        v16h af[2], bf[2];
        for (int sm = 0; sm < 2; ++sm) {
            int row = wm * 32 + sm * 16 + lm;
            // A layout: lane half lh supplies contiguous k range [16*lh, 16*lh+16)
            v8h lo = *(const v8h*)&As[row * 32 + 16 * lh];
            v8h hi = *(const v8h*)&As[row * 32 + 16 * lh + 8];
            for (int e = 0; e < 8; ++e) { af[sm][e] = lo[e]; af[sm][e + 8] = hi[e]; }
        }
        for (int sn = 0; sn < 2; ++sn) {
            int col = wn * 32 + sn * 16 + lm;
            // B layout: lane half lh supplies contiguous k range [16*lh, 16*lh+16)
            bf[sn] = *(const v16h*)&Bt[col * 32 + 16 * lh];
        }
        for (int sm = 0; sm < 2; ++sm)
            for (int sn = 0; sn < 2; ++sn)
                acc[sm][sn] = wmma_f16(af[sm], bf[sn], acc[sm][sn]);
        __syncthreads();
    }

    // --- epilogue: C layout: row = r + 8*lh, col = lm ---
    for (int sm = 0; sm < 2; ++sm)
        for (int sn = 0; sn < 2; ++sn)
            for (int r = 0; r < 8; ++r) {
                int row = m0 + wm * 32 + sm * 16 + r + 8 * lh;
                int col = n0 + wn * 32 + sn * 16 + lm;
                if (col < N) {
                    float v = acc[sm][sn][r];
                    if (BIAS)   v += bias[col];
                    if (RES)    v += res[(size_t)row * ldo + col];
                    if (DOGELU) v = 0.5f * v * (1.0f + erff(v * 0.70710678118654752f));
                    if (OUTF16) out16[(size_t)row * ldo + col] = (_Float16)v;
                    else        out32[(size_t)row * ldo + col] = v;
                }
            }
}

// ---------------------------------------------------------------------------
// Causal attention, one wave (32 threads) per (query-tile-of-16, head, batch).
// qkv layout: [B*T, 1536] f16 with q|k|v each [H=8][D=64].
// Dynamic LDS: scores 16 x 1024 f32 = 64 KB.
__global__ __launch_bounds__(32)
void attn_kernel(const _Float16* __restrict__ qkv, _Float16* __restrict__ y16) {
    extern __shared__ char smem_dyn[];
    float* sc = (float*)smem_dyn;          // [16][1024]

    int qt = blockIdx.x;                   // 0..63 query tiles
    int h  = blockIdx.y;                   // 0..7
    int b  = blockIdx.z;                   // 0..1
    int lane = threadIdx.x;
    int lh = lane >> 4, lm = lane & 15;
    int qbase = qt * 16;

    const _Float16* qp = qkv + (size_t)b * 1024 * 1536 + (size_t)h * 64;
    const _Float16* kp = qp + 512;
    const _Float16* vp = qp + 1024;

    // Q A-fragments (K-dim = head_dim 64 -> 2 fragments of k=32)
    v16h aq[2];
    for (int f = 0; f < 2; ++f) {
        int d0 = f * 32 + 16 * lh;
        const _Float16* qrow = qp + (size_t)(qbase + lm) * 1536;
        v8h lo = *(const v8h*)(qrow + d0);
        v8h hi = *(const v8h*)(qrow + d0 + 8);
        for (int e = 0; e < 8; ++e) { aq[f][e] = lo[e]; aq[f][e + 8] = hi[e]; }
    }

    int ntiles = qt + 1;                   // causal: key tiles 0..qt
    // ---- Phase 1: S = Q K^T, scaled + masked, into LDS ----
    for (int j = 0; j < ntiles; ++j) {
        int tk = j * 16 + lm;              // key index for this lane's column
        v16h bk[2];
        for (int f = 0; f < 2; ++f) {
            int d0 = f * 32 + 16 * lh;     // B frag: contiguous k (=d) run
            bk[f] = *(const v16h*)(kp + (size_t)tk * 1536 + d0);
        }
        v8f c;
        for (int e = 0; e < 8; ++e) c[e] = 0.0f;
        c = wmma_f16(aq[0], bk[0], c);
        c = wmma_f16(aq[1], bk[1], c);
        for (int r = 0; r < 8; ++r) {
            int Mq  = r + 8 * lh;
            int key = j * 16 + lm;
            float s = c[r] * 0.125f;       // 1/sqrt(64)
            if (key > qbase + Mq) s = -1e30f;   // causal mask
            sc[Mq * 1024 + key] = s;
        }
    }
    __syncthreads();

    int ncols  = ntiles * 16;
    int ksteps = (ncols + 31) / 32;
    int npad   = ksteps * 32;

    // ---- softmax per row (lanes 0..15, one row each) ----
    if (lane < 16) {
        int row = lane;
        float* sr = sc + row * 1024;
        float mx = -1e30f;
        for (int i = 0; i < ncols; ++i) mx = fmaxf(mx, sr[i]);
        float sum = 0.0f;
        for (int i = 0; i < ncols; ++i) {
            float e = expf(sr[i] - mx);
            sr[i] = e;
            sum += e;
        }
        float inv = 1.0f / sum;
        for (int i = 0; i < ncols; ++i) sr[i] *= inv;
        for (int i = ncols; i < npad; ++i) sr[i] = 0.0f;  // pad to k-step
    }
    __syncthreads();

    // ---- Phase 2: O = P V ----
    v8f o[4];
    for (int dn = 0; dn < 4; ++dn)
        for (int e = 0; e < 8; ++e) o[dn][e] = 0.0f;

    for (int kt = 0; kt < ksteps; ++kt) {
        // A frag from probabilities (f32 LDS -> f16)
        v16h ap;
        for (int jj = 0; jj < 16; ++jj) {
            int k = kt * 32 + jj + 16 * lh;
            ap[jj] = (_Float16)sc[lm * 1024 + k];
        }
        for (int dn = 0; dn < 4; ++dn) {
            v16h bv;
            for (int e = 0; e < 16; ++e) {
                int key = kt * 32 + e + 16 * lh;
                bv[e] = vp[(size_t)key * 1536 + dn * 16 + lm];
            }
            o[dn] = wmma_f16(ap, bv, o[dn]);
        }
    }

    // store O as f16 into y16 [B*T, 512] at (h*64 + d)
    for (int dn = 0; dn < 4; ++dn)
        for (int r = 0; r < 8; ++r) {
            int Mq = r + 8 * lh;
            y16[(size_t)(b * 1024 + qbase + Mq) * 512 + h * 64 + dn * 16 + lm] =
                (_Float16)o[dn][r];
        }
}

// ---------------------------------------------------------------------------
extern "C" void kernel_launch(void* const* d_in, const int* in_sizes, int n_in,
                              void* d_out, int out_size, void* d_ws, size_t ws_size,
                              hipStream_t stream) {
    (void)in_sizes; (void)n_in; (void)out_size; (void)ws_size;
    const int*   idx    = (const int*)  d_in[0];
    const float* wte    = (const float*)d_in[1];
    const float* wpe    = (const float*)d_in[2];
    const float* ln1_g  = (const float*)d_in[3];
    const float* ln1_b  = (const float*)d_in[4];
    const float* attn_w = (const float*)d_in[5];
    const float* attn_b = (const float*)d_in[6];
    const float* proj_w = (const float*)d_in[7];
    const float* proj_b = (const float*)d_in[8];
    const float* ln2_g  = (const float*)d_in[9];
    const float* ln2_b  = (const float*)d_in[10];
    const float* fc_w   = (const float*)d_in[11];
    const float* fc_b   = (const float*)d_in[12];
    const float* fc2_w  = (const float*)d_in[13];
    const float* fc2_b  = (const float*)d_in[14];
    const float* lnf_g  = (const float*)d_in[15];
    const float* lnf_b  = (const float*)d_in[16];
    float* out = (float*)d_out;

    char* base = (char*)d_ws;
    size_t off = 0;
    auto carve = [&](size_t bytes) -> void* {
        void* p = base + off;
        off += (bytes + 255) & ~(size_t)255;
        return p;
    };
    float*    x     = (float*)   carve(2048ull * 512 * 4);   // residual stream
    _Float16* h16   = (_Float16*)carve(2048ull * 512 * 2);   // LN output
    _Float16* qkv16 = (_Float16*)carve(2048ull * 1536 * 2);  // q|k|v
    _Float16* y16   = (_Float16*)carve(2048ull * 512 * 2);   // attention out
    _Float16* u16   = (_Float16*)carve(2048ull * 2048 * 2);  // MLP hidden

    const int V = 50257, E = 512;

    // x = wte[idx] + wpe
    embed_kernel<<<2048, 256, 0, stream>>>(idx, wte, wpe, x);

    for (int l = 0; l < 6; ++l) {
        // h = LN1(x)
        ln_kernel<<<2048, 256, 0, stream>>>(x, ln1_g + l * E, ln1_b + l * E, h16);
        // qkv = h @ attn_w + attn_b   (f16 out)
        gemm_kernel<false, false, true, false, true>
            <<<dim3(1536 / 64, 2048 / 64), 128, 0, stream>>>(
                h16, E, attn_w + (size_t)l * E * 1536, 1536,
                attn_b + (size_t)l * 1536, nullptr, nullptr, qkv16,
                1536, 2048, 1536, E);
        // causal multi-head attention -> y16
        attn_kernel<<<dim3(64, 8, 2), 32, 16 * 1024 * 4, stream>>>(qkv16, y16);
        // x = x + y @ proj_w + proj_b
        gemm_kernel<false, false, false, true, true>
            <<<dim3(512 / 64, 2048 / 64), 128, 0, stream>>>(
                y16, E, proj_w + (size_t)l * E * E, E,
                proj_b + (size_t)l * E, x, x, nullptr,
                E, 2048, E, E);
        // h = LN2(x)
        ln_kernel<<<2048, 256, 0, stream>>>(x, ln2_g + l * E, ln2_b + l * E, h16);
        // u = gelu(h @ fc_w + fc_b)   (f16 out)
        gemm_kernel<false, true, true, false, true>
            <<<dim3(2048 / 64, 2048 / 64), 128, 0, stream>>>(
                h16, E, fc_w + (size_t)l * E * 2048, 2048,
                fc_b + (size_t)l * 2048, nullptr, nullptr, u16,
                2048, 2048, 2048, E);
        // x = x + u @ fc2_w + fc2_b
        gemm_kernel<false, false, false, true, true>
            <<<dim3(512 / 64, 2048 / 64), 128, 0, stream>>>(
                u16, 2048, fc2_w + (size_t)l * 2048 * E, E,
                fc2_b + (size_t)l * E, x, x, nullptr,
                E, 2048, E, 2048);
    }

    // h = LN_f(x)
    ln_kernel<<<2048, 256, 0, stream>>>(x, lnf_g, lnf_b, h16);
    // logits = h @ wte^T   (B given as [V,E] -> BT path), f32 out
    gemm_kernel<true, false, false, false, false>
        <<<dim3((V + 63) / 64, 2048 / 64), 128, 0, stream>>>(
            h16, E, wte, E, nullptr, nullptr, out, nullptr,
            V, 2048, V, E);
}
